// LNNODE_29154238005968
// MI455X (gfx1250) — matrix operand order repaired and checked by reading, not compiled, not run on hardware
//
#include <hip/hip_runtime.h>
#include <hip/hip_bf16.h>

typedef __attribute__((ext_vector_type(16))) _Float16 v16h;
typedef __attribute__((ext_vector_type(8)))  _Float16 v8h;
typedef __attribute__((ext_vector_type(8)))  float    v8f;

#define HID   256
#define BT    16          // batch elements per workgroup
#define NROWS 112         // 7 row-types (fwd + 6 tangents) * 16 elements
#define RSTR  264         // LDS row stride in halves (256 + 8 pad -> no bank conflicts)
#define NTRI  10
#define QD    6
#define DQD   4
#define XDIM  10
#define EPSI  1e-3f

// ---------------------------------------------------------------------------
// Prep kernel: pack W (K=256 rows, N cols, f32 row-major) into WMMA B-matrix
// fragments (f16). Fragment layout this code uses for v_wmma_f32_16x16x32_f16 B (32x16):
//   lane 0..15  -> column N = lane,    K = kt*32 + 0..15  (16 contiguous halves)
//   lane 16..31 -> column N = lane-16, K = kt*32 + 16..31
// stored fragment-major so each lane's 16 halves are 32 contiguous bytes.
// ---------------------------------------------------------------------------
__global__ void pack_b_kernel(const float* __restrict__ W, _Float16* __restrict__ out,
                              int N, int ntiles) {
  int idx = blockIdx.x * blockDim.x + threadIdx.x;
  int total = ntiles * 8 * 32 * 16;
  if (idx >= total) return;
  int h    = idx & 15;
  int lane = (idx >> 4) & 31;
  int kt   = (idx >> 9) & 7;
  int nt   = idx >> 12;
  int n = nt * 16 + (lane & 15);
  int k = kt * 32 + ((lane & 16) ? 16 : 0) + h;
  float v = (n < N) ? W[k * N + n] : 0.0f;
  out[idx] = (_Float16)v;
}

// ---------------------------------------------------------------------------
// One 16x16 output tile of act(M-rows) @ Wpk, K=256 (8 WMMA steps).
// A fragment (contiguous-k order, same as packed B): lane m = lane&15, halves are two 8-K
// chunks at K = kt*32 + base(+0) and + base+8, base = (lane<16 ? 0 : 16).
// ---------------------------------------------------------------------------
__device__ __forceinline__ v8f wmma_row_tile(const _Float16* act,
                                             const _Float16* __restrict__ Bpk,
                                             int mtile, int ntile, int lane) {
  v8f c = {};
  const _Float16* arow = act + (mtile * 16 + (lane & 15)) * RSTR;
  const int alo = (lane & 16) ? 16 : 0;
  const _Float16* bp = Bpk + (size_t)(ntile * 8) * 32 * 16 + lane * 16;
#pragma unroll
  for (int kt = 0; kt < 8; ++kt) {
    v8h a0 = *(const v8h*)(arow + kt * 32 + alo);
    v8h a1 = *(const v8h*)(arow + kt * 32 + alo + 8);
    v16h a;
#pragma unroll
    for (int i = 0; i < 8; ++i) { a[i] = a0[i]; a[i + 8] = a1[i]; }
    v16h b = *(const v16h*)bp;
    if (kt < 7) __builtin_prefetch((const void*)(bp + 32 * 16), 0, 1);
    c = __builtin_amdgcn_wmma_f32_16x16x32_f16(false, a, false, b, (short)0, c,
                                               false, false);
    bp += 32 * 16;
  }
  return c;
}

// ---------------------------------------------------------------------------
// Layer 1 + tangent seeding: z1 = q@W0+b0 ; h1 = tanh ; rows j=0 -> h1,
// rows j=1..6 -> W0[j-1,:] * (1-h1^2)   (JVP of tanh layer for e_k directions)
// 128 threads: thread = (e, 32-feature chunk).
// ---------------------------------------------------------------------------
__device__ __forceinline__ void layer1_tangents(const float* __restrict__ W0,
                                                const float* __restrict__ b0,
                                                const float* xt, _Float16* act,
                                                int tid) {
  const int e  = tid >> 3;
  const int f0 = (tid & 7) * 32;
  float q[QD];
#pragma unroll
  for (int d = 0; d < QD; ++d) q[d] = xt[e * XDIM + d];
  for (int i = 0; i < 32; ++i) {
    const int f = f0 + i;
    float w[QD];
    float z = b0[f];
#pragma unroll
    for (int d = 0; d < QD; ++d) { w[d] = W0[d * HID + f]; z += q[d] * w[d]; }
    const float h1 = tanhf(z);
    const float d1 = 1.0f - h1 * h1;
    act[(0 * BT + e) * RSTR + f] = (_Float16)h1;
#pragma unroll
    for (int k = 0; k < QD; ++k)
      act[((k + 1) * BT + e) * RSTR + f] = (_Float16)(w[k] * d1);
  }
}

// ---------------------------------------------------------------------------
// Layer 2: pass A = forward rows (tanh, record d2), pass B = tangent rows * d2
// ---------------------------------------------------------------------------
__device__ __forceinline__ void layer2(const _Float16* actIn, _Float16* actOut,
                                       const _Float16* __restrict__ W1pk,
                                       const float* __restrict__ b1,
                                       float* d2buf, int wave, int lane) {
  const int eb = (lane & 16) ? 8 : 0;
#pragma unroll
  for (int t = 0; t < 4; ++t) {             // forward: mtile 0, 16 N-tiles / 4 waves
    const int nt = wave * 4 + t;
    v8f c = wmma_row_tile(actIn, W1pk, 0, nt, lane);
    const int n = nt * 16 + (lane & 15);
    const float bn = b1[n];
#pragma unroll
    for (int v = 0; v < 8; ++v) {
      const int e = eb + v;                 // C/D layout: M = v + 8*(lane>=16)
      const float h2 = tanhf(c[v] + bn);
      actOut[(0 * BT + e) * RSTR + n] = (_Float16)h2;
      d2buf[e * HID + n] = 1.0f - h2 * h2;
    }
  }
  __syncthreads();
  for (int t = 0; t < 24; ++t) {            // tangents: mtiles 1..6 x 16 N-tiles
    const int g  = wave * 24 + t;
    const int mt = 1 + (g >> 4);
    const int nt = g & 15;
    v8f c = wmma_row_tile(actIn, W1pk, mt, nt, lane);
    const int n = nt * 16 + (lane & 15);
#pragma unroll
    for (int v = 0; v < 8; ++v) {
      const int e = eb + v;
      actOut[(mt * BT + e) * RSTR + n] = (_Float16)(c[v] * d2buf[e * HID + n]);
    }
  }
  __syncthreads();
}

__device__ __forceinline__ float softplus_f(float v) {
  return (v > 20.0f) ? v : log1pf(expf(v));
}
__device__ __forceinline__ float sigmoid_f(float v) {
  return 1.0f / (1.0f + expf(-v));
}

// ---------------------------------------------------------------------------
// Main fused dynamics kernel: 128 threads (4 wave32), 16 elements / block.
// ---------------------------------------------------------------------------
__global__ void __launch_bounds__(128, 1)
lnn_dynamics_kernel(const float* __restrict__ x,
                    const float* __restrict__ mW0, const float* __restrict__ mb0,
                    const _Float16* __restrict__ mW1pk, const float* __restrict__ mb1,
                    const _Float16* __restrict__ mW2pk, const float* __restrict__ mb2,
                    const float* __restrict__ vW0, const float* __restrict__ vb0,
                    const _Float16* __restrict__ vW1pk, const float* __restrict__ vb1,
                    const _Float16* __restrict__ vW2pk,
                    float* __restrict__ out) {
  __shared__ __align__(16) _Float16 actA[NROWS * RSTR];   // 59,136 B
  __shared__ __align__(16) _Float16 actB[NROWS * RSTR];   // 59,136 B
  __shared__ float d2buf[BT * HID];                       // 16 KB
  __shared__ float xt[BT * XDIM];
  __shared__ float entm[BT * 16];
  __shared__ float tentm[6 * BT * 16];
  __shared__ float tentv[6 * BT * 16];

  const int tid  = threadIdx.x;
  const int lane = tid & 31;
  const int wave = tid >> 5;
  const int base = blockIdx.x * BT;
  const int eb   = (lane & 16) ? 8 : 0;

  for (int i = tid; i < BT * XDIM; i += 128) xt[i] = x[base * XDIM + i];
  __syncthreads();

  // ======================= MASS MLP ==========================
  layer1_tangents(mW0, mb0, xt, actA, tid);
  __syncthreads();
  layer2(actA, actB, mW1pk, mb1, d2buf, wave, lane);

  // layer 3: 256 -> 10 (padded N-tile), mtiles 0..6
  for (int mt = wave; mt < 7; mt += 4) {
    v8f c = wmma_row_tile(actB, mW2pk, mt, 0, lane);
    const int n = lane & 15;
#pragma unroll
    for (int v = 0; v < 8; ++v) {
      const int e = eb + v;
      if (mt == 0) {
        entm[e * 16 + n] = c[v] + ((n < NTRI) ? mb2[n] : 0.0f);
      } else {
        tentm[((mt - 1) * BT + e) * 16 + n] = c[v];
      }
    }
  }
  __syncthreads();

  // ======================== V MLP ============================
  layer1_tangents(vW0, vb0, xt, actA, tid);
  __syncthreads();
  layer2(actA, actB, vW1pk, vb1, d2buf, wave, lane);

  // layer 3 (V): only tangent rows needed (dV/dq_k); value of V cancels.
  for (int g = wave; g < 6; g += 4) {
    const int mt = g + 1;
    v8f c = wmma_row_tile(actB, vW2pk, mt, 0, lane);
    const int n = lane & 15;
#pragma unroll
    for (int v = 0; v < 8; ++v) {
      const int e = eb + v;
      tentv[(g * BT + e) * 16 + n] = c[v];
    }
  }
  __syncthreads();

  // ======================= head math =========================
  if (tid < BT) {
    const int e = tid;
    const int TI[NTRI] = {0, 1, 1, 2, 2, 2, 3, 3, 3, 3};
    const int TJ[NTRI] = {0, 0, 1, 0, 1, 2, 0, 1, 2, 3};

    float qv[QD], dq[DQD];
#pragma unroll
    for (int d = 0; d < QD; ++d) qv[d] = xt[e * XDIM + d];
#pragma unroll
    for (int d = 0; d < DQD; ++d) dq[d] = xt[e * XDIM + QD + d];
    const float x1[2]  = {qv[2], qv[3]};
    const float x2[2]  = {qv[4], qv[5]};
    const float dth[2] = {dq[2], dq[3]};

    // L (lower), softplus on diagonal; sigmoid = softplus' for tangent chain
    float sig[NTRI];
    float L[DQD][DQD] = {};
#pragma unroll
    for (int t = 0; t < NTRI; ++t) {
      const float v = entm[e * 16 + t];
      if (TI[t] == TJ[t]) { sig[t] = sigmoid_f(v); L[TI[t]][TJ[t]] = softplus_f(v); }
      else                { sig[t] = 1.0f;         L[TI[t]][TJ[t]] = v; }
    }
    float M[DQD][DQD];
#pragma unroll
    for (int i = 0; i < DQD; ++i)
#pragma unroll
      for (int j = 0; j < DQD; ++j) {
        float s = 0.0f;
#pragma unroll
        for (int k = 0; k < DQD; ++k) s += L[i][k] * L[j][k];
        M[i][j] = s + ((i == j) ? EPSI : 0.0f);
      }

    // J columns: J[:,k] = (dL_k L^T + L dL_k^T) dq
    float Jc[QD][DQD];
#pragma unroll
    for (int k = 0; k < QD; ++k) {
      float dL[DQD][DQD] = {};
#pragma unroll
      for (int t = 0; t < NTRI; ++t)
        dL[TI[t]][TJ[t]] = tentm[(k * BT + e) * 16 + t] * sig[t];
#pragma unroll
      for (int i = 0; i < DQD; ++i) {
        float acc = 0.0f;
#pragma unroll
        for (int j = 0; j < DQD; ++j) {
          float dm = 0.0f;
#pragma unroll
          for (int c = 0; c < DQD; ++c)
            dm += dL[i][c] * L[j][c] + L[i][c] * dL[j][c];
          acc += dm * dq[j];
        }
        Jc[k][i] = acc;
      }
    }

    // dqMdq: [J[:,0:2] | -x2*J[:,2:4] + x1*J[:,4:6]]
    float dqMdq[DQD][DQD];
#pragma unroll
    for (int i = 0; i < DQD; ++i) {
      dqMdq[i][0] = Jc[0][i];
      dqMdq[i][1] = Jc[1][i];
      dqMdq[i][2] = -x2[0] * Jc[2][i] + x1[0] * Jc[4][i];
      dqMdq[i][3] = -x2[1] * Jc[3][i] + x1[1] * Jc[5][i];
    }

    // gradL wrt q: 0.5*dq.J[:,k] - dV/dq_k   (V forward value cancels)
    float gl[QD];
#pragma unroll
    for (int k = 0; k < QD; ++k) {
      float s = 0.0f;
#pragma unroll
      for (int i = 0; i < DQD; ++i) s += dq[i] * Jc[k][i];
      gl[k] = 0.5f * s - tentv[(k * BT + e) * 16 + 0];
    }
    const float dLdq[DQD] = {gl[0], gl[1],
                             -x2[0] * gl[2] + x1[0] * gl[4],
                             -x2[1] * gl[3] + x1[1] * gl[5]};

    float r[DQD];
#pragma unroll
    for (int i = 0; i < DQD; ++i) {
      float s = 0.0f;
#pragma unroll
      for (int j = 0; j < DQD; ++j) s += dqMdq[i][j] * dq[j];
      r[i] = dLdq[i] - s;
    }

    // 4x4 Cholesky of M, then solve M ddq = r
    float C[DQD][DQD] = {};
#pragma unroll
    for (int j = 0; j < DQD; ++j) {
      float s = M[j][j];
      for (int k = 0; k < j; ++k) s -= C[j][k] * C[j][k];
      C[j][j] = sqrtf(s);
      const float inv = 1.0f / C[j][j];
      for (int i = j + 1; i < DQD; ++i) {
        float t2 = M[i][j];
        for (int k = 0; k < j; ++k) t2 -= C[i][k] * C[j][k];
        C[i][j] = t2 * inv;
      }
    }
    float y[DQD];
#pragma unroll
    for (int i = 0; i < DQD; ++i) {
      float s = r[i];
      for (int k = 0; k < i; ++k) s -= C[i][k] * y[k];
      y[i] = s / C[i][i];
    }
    float dd[DQD];
#pragma unroll
    for (int ii = DQD - 1; ii >= 0; --ii) {
      float s = y[ii];
      for (int k = ii + 1; k < DQD; ++k) s -= C[k][ii] * dd[k];
      dd[ii] = s / C[ii][ii];
    }

    float* op = out + (size_t)(base + e) * XDIM;
    op[0] = dq[0];
    op[1] = dq[1];
    op[2] = -x2[0] * dth[0];
    op[3] = -x2[1] * dth[1];
    op[4] =  x1[0] * dth[0];
    op[5] =  x1[1] * dth[1];
    op[6] = dd[0];
    op[7] = dd[1];
    op[8] = dd[2];
    op[9] = dd[3];
  }
}

// ---------------------------------------------------------------------------
extern "C" void kernel_launch(void* const* d_in, const int* in_sizes, int n_in,
                              void* d_out, int out_size, void* d_ws, size_t ws_size,
                              hipStream_t stream) {
  const float* x   = (const float*)d_in[0];
  const float* mW0 = (const float*)d_in[1];
  const float* mb0 = (const float*)d_in[2];
  const float* mW1 = (const float*)d_in[3];
  const float* mb1 = (const float*)d_in[4];
  const float* mW2 = (const float*)d_in[5];
  const float* mb2 = (const float*)d_in[6];
  const float* vW0 = (const float*)d_in[7];
  const float* vb0 = (const float*)d_in[8];
  const float* vW1 = (const float*)d_in[9];
  const float* vb1 = (const float*)d_in[10];
  const float* vW2 = (const float*)d_in[11];
  // vb2 (d_in[12]) is mathematically irrelevant: only dV/dq is consumed.

  _Float16* ws    = (_Float16*)d_ws;
  _Float16* mW1pk = ws;             // 16 N-tiles * 8 K-tiles * 512 halves = 65536
  _Float16* vW1pk = ws + 65536;
  _Float16* mW2pk = ws + 131072;    // 1 N-tile (10 cols padded to 16) = 4096
  _Float16* vW2pk = ws + 135168;    // 1 N-tile (1 col padded)         = 4096

  pack_b_kernel<<<256, 256, 0, stream>>>(mW1, mW1pk, HID, 16);
  pack_b_kernel<<<256, 256, 0, stream>>>(vW1, vW1pk, HID, 16);
  pack_b_kernel<<<16, 256, 0, stream>>>(mW2, mW2pk, NTRI, 1);
  pack_b_kernel<<<16, 256, 0, stream>>>(vW2, vW2pk, 1, 1);

  const int bs = in_sizes[0] / XDIM;
  lnn_dynamics_kernel<<<bs / BT, 128, 0, stream>>>(
      x, mW0, mb0, mW1pk, mb1, mW2pk, mb2, vW0, vb0, vW1pk, vb1, vW2pk,
      (float*)d_out);
}
